// GraphAttentionLayerV2_50663434224276
// MI455X (gfx1250) — matrix operand order repaired and checked by reading, not compiled, not run on hardware
//
#include <hip/hip_runtime.h>
#include <hip/hip_bf16.h>

#define NROW 8192
#define CDIM 512
#define GAT_ALPHA 0.2f
#define NEG_BIG -9e15f

typedef __attribute__((ext_vector_type(16))) __bf16 bf16x16;
typedef __attribute__((ext_vector_type(8)))  float  f32x8;
typedef __attribute__((ext_vector_type(4)))  unsigned int u32x4;

struct U32x8 { u32x4 lo, hi; };

static __device__ __forceinline__ bf16x16 make_frag(u32x4 lo, u32x4 hi) {
    U32x8 t; t.lo = lo; t.hi = hi;
    return __builtin_bit_cast(bf16x16, t);
}

// round-to-nearest-even f32 -> bf16 (raw bits)
static __device__ __forceinline__ unsigned short f2bf(float f) {
    unsigned int x = __float_as_uint(f);
    x += 0x7fffu + ((x >> 16) & 1u);
    return (unsigned short)(x >> 16);
}

// ---------------------------------------------------------------------------
// K1: w1a[k] = sum_c W[k][c]*a_w[c] ; w2a[k] = sum_c W[512+k][c]*a_w[c]
// one wave per k (1024 waves)
// ---------------------------------------------------------------------------
__global__ __launch_bounds__(256) void k_prep_wa(const float* __restrict__ W,
                                                 const float* __restrict__ a_w,
                                                 float* __restrict__ w1a,
                                                 float* __restrict__ w2a) {
    int wave = (blockIdx.x * blockDim.x + threadIdx.x) >> 5;
    int lane = threadIdx.x & 31;
    if (wave >= 2 * CDIM) return;
    const float* row = W + (size_t)wave * CDIM;
    float a = 0.f;
    for (int c = lane; c < CDIM; c += 32) a += row[c] * a_w[c];
    #pragma unroll
    for (int off = 16; off; off >>= 1) a += __shfl_xor(a, off, 32);
    if (lane == 0) {
        if (wave < CDIM) w1a[wave] = a;
        else             w2a[wave - CDIM] = a;
    }
}

// ---------------------------------------------------------------------------
// K2: h (f32) -> h_bf16, 4 elems/thread
// ---------------------------------------------------------------------------
__global__ __launch_bounds__(256) void k_cvt_h(const float* __restrict__ h,
                                               unsigned int* __restrict__ hb) {
    size_t i = (size_t)(blockIdx.x * 256 + threadIdx.x) * 4;
    const float4 v = *reinterpret_cast<const float4*>(h + i);
    unsigned int lo = (unsigned)f2bf(v.x) | ((unsigned)f2bf(v.y) << 16);
    unsigned int hi = (unsigned)f2bf(v.z) | ((unsigned)f2bf(v.w) << 16);
    hb[i / 2]     = lo;
    hb[i / 2 + 1] = hi;
}

// ---------------------------------------------------------------------------
// K3: fc_w [k][c] f32 -> fcwT [c][k] bf16 (transposed so WMMA B loads are
// contiguous in K)
// ---------------------------------------------------------------------------
__global__ __launch_bounds__(256) void k_cvt_fcwT(const float* __restrict__ fcw,
                                                  unsigned short* __restrict__ fcwT) {
    int idx = blockIdx.x * 256 + threadIdx.x;   // 512*512
    int k = idx >> 9, c = idx & 511;
    fcwT[(size_t)c * CDIM + k] = f2bf(fcw[idx]);
}

// ---------------------------------------------------------------------------
// K4: s1[i] = h[i]·w1a ; s2[i] = h[i]·w2a ; one wave per row
// ---------------------------------------------------------------------------
__global__ __launch_bounds__(256) void k_s12(const float* __restrict__ h,
                                             const float* __restrict__ w1a,
                                             const float* __restrict__ w2a,
                                             float* __restrict__ s1,
                                             float* __restrict__ s2) {
    int wave = (blockIdx.x * blockDim.x + threadIdx.x) >> 5;
    int lane = threadIdx.x & 31;
    if (wave >= NROW) return;
    const float* row = h + (size_t)wave * CDIM;
    float a0 = 0.f, a1 = 0.f;
    for (int c = lane; c < CDIM; c += 32) {
        float hv = row[c];
        a0 += hv * w1a[c];
        a1 += hv * w2a[c];
    }
    #pragma unroll
    for (int off = 16; off; off >>= 1) {
        a0 += __shfl_xor(a0, off, 32);
        a1 += __shfl_xor(a1, off, 32);
    }
    if (lane == 0) { s1[wave] = a0; s2[wave] = a1; }
}

// ---------------------------------------------------------------------------
// K5: hfcT[c][j] = bf16( (h @ fc_w)[j][c] + fc_b[c] )
// grid (N/128, C/64); 8 waves; wave w owns rows [bm*128+w*16, +16), 64 cols.
// ---------------------------------------------------------------------------
__global__ __launch_bounds__(256) void k_hfc_gemm(const unsigned short* __restrict__ hb,
                                                  const unsigned short* __restrict__ fcwT,
                                                  const float* __restrict__ fc_b,
                                                  unsigned short* __restrict__ hfcT) {
    const int wave = threadIdx.x >> 5, lane = threadIdx.x & 31;
    const int row0 = blockIdx.x * 128 + wave * 16;
    const int col0 = blockIdx.y * 64;
    const int m    = lane & 15;
    const int half = lane >> 4;           // 0: lanes 0-15, 1: lanes 16-31
    const int kloA = half * 8;            // A K-chunk base
    const int kloB = half * 8;            // B K base

    f32x8 acc[4] = {};

    const unsigned short* arow = hb + (size_t)(row0 + m) * CDIM;
    for (int k = 0; k < CDIM; k += 32) {
        u32x4 alo = *reinterpret_cast<const u32x4*>(arow + k + kloA);
        u32x4 ahi = *reinterpret_cast<const u32x4*>(arow + k + kloA + 16);
        bf16x16 afrag = make_frag(alo, ahi);
        #pragma unroll
        for (int nt = 0; nt < 4; ++nt) {
            const unsigned short* bp =
                fcwT + (size_t)(col0 + nt * 16 + m) * CDIM + k + kloB;
            bf16x16 bfrag = make_frag(*reinterpret_cast<const u32x4*>(bp),
                                      *reinterpret_cast<const u32x4*>(bp + 16));
            acc[nt] = __builtin_amdgcn_wmma_f32_16x16x32_bf16(
                false, afrag, false, bfrag, (short)0, acc[nt], false, false);
        }
    }

    // epilogue: + bias, store transposed as bf16 (8 consecutive rows per lane)
    const int rstore = row0 + half * 8;
    #pragma unroll
    for (int nt = 0; nt < 4; ++nt) {
        int col = col0 + nt * 16 + m;
        float bias = fc_b[col];
        u32x4 o;
        #pragma unroll
        for (int p = 0; p < 4; ++p) {
            unsigned short e0 = f2bf(acc[nt][2 * p]     + bias);
            unsigned short e1 = f2bf(acc[nt][2 * p + 1] + bias);
            o[p] = (unsigned)e0 | ((unsigned)e1 << 16);
        }
        *reinterpret_cast<u32x4*>(hfcT + (size_t)col * NROW + rstore) = o;
    }
}

// ---------------------------------------------------------------------------
// K6: per-row online softmax stats (max, sum) over
//     e = mask(lrelu(s1[i]+s2[j]+a_b)); one block per row
// ---------------------------------------------------------------------------
__global__ __launch_bounds__(256) void k_row_stats(const float* __restrict__ adj,
                                                   const float* __restrict__ s1,
                                                   const float* __restrict__ s2,
                                                   const float* __restrict__ a_b,
                                                   float* __restrict__ rowmax,
                                                   float* __restrict__ rowsum) {
    const int i   = blockIdx.x;
    const int tid = threadIdx.x;
    const float base = s1[i] + a_b[0];
    const float* arow = adj + (size_t)i * NROW;

    float m = -INFINITY, l = 0.f;
    for (int j = tid; j < NROW; j += 256) {
        float e = base + s2[j];
        e = (e >= 0.f) ? e : GAT_ALPHA * e;
        e = (arow[j] > 0.f) ? e : NEG_BIG;      // e always finite
        if (e > m) { l = l * __expf(m - e) + 1.f; m = e; }
        else       { l += __expf(e - m); }
    }

    __shared__ float sm[256], sl[256];
    sm[tid] = m; sl[tid] = l;
    __syncthreads();
    for (int s = 128; s; s >>= 1) {
        if (tid < s) {
            float m1 = sm[tid], m2 = sm[tid + s];
            float M = fmaxf(m1, m2);
            sl[tid] = sl[tid] * __expf(m1 - M) + sl[tid + s] * __expf(m2 - M);
            sm[tid] = M;
        }
        __syncthreads();
    }
    if (tid == 0) { rowmax[i] = sm[0]; rowsum[i] = sl[0]; }
}

// ---------------------------------------------------------------------------
// K7: out = softmax(e) @ hfc, fused: P tile regenerated from s1/s2/adj,
// exp'd, bf16'd into LDS, then WMMA against hfcT.
// grid 256 blocks; block = 32 rows x all 512 cols; wave w owns 64 cols.
// ---------------------------------------------------------------------------
__global__ __launch_bounds__(256) void k_att_gemm(const float* __restrict__ adj,
                                                  const float* __restrict__ s1v,
                                                  const float* __restrict__ s2v,
                                                  const float* __restrict__ a_b,
                                                  const float* __restrict__ rowmax,
                                                  const float* __restrict__ rowsum,
                                                  const unsigned short* __restrict__ hfcT,
                                                  float* __restrict__ out) {
    const int tid  = threadIdx.x;
    const int wave = tid >> 5, lane = tid & 31;
    const int row0 = blockIdx.x * 32;
    const int m    = lane & 15;
    const int half = lane >> 4;
    const int col0 = wave * 64;

    __shared__ unsigned short sP[32 * 32];   // P tile, bf16
    __shared__ float sS1[32], sRM[32], sInvL[32];
    if (tid < 32) {
        sS1[tid]  = s1v[row0 + tid] + a_b[0];
        sRM[tid]  = rowmax[row0 + tid];
        sInvL[tid] = 1.f / rowsum[row0 + tid];
    }
    __syncthreads();

    // P-producer registers: this thread fills rows pr+8t, col pjj
    const int pjj = tid & 31;
    const int pr  = tid >> 5;                // 0..7
    float rs1[4], rrm[4];
    const float* arow[4];
    #pragma unroll
    for (int t = 0; t < 4; ++t) {
        int r = pr + 8 * t;
        rs1[t]  = sS1[r];
        rrm[t]  = sRM[r];
        arow[t] = adj + (size_t)(row0 + r) * NROW;
    }

    f32x8 acc[2][4] = {};

    for (int j0 = 0; j0 < NROW; j0 += 32) {
        // ---- produce P[32][32] = exp(e - rowmax) as bf16 in LDS ----
        float s2j = s2v[j0 + pjj];
        #pragma unroll
        for (int t = 0; t < 4; ++t) {
            int r = pr + 8 * t;
            float e = rs1[t] + s2j;
            e = (e >= 0.f) ? e : GAT_ALPHA * e;
            float av = arow[t][j0 + pjj];
            e = (av > 0.f) ? e : NEG_BIG;
            sP[r * 32 + pjj] = f2bf(__expf(e - rrm[t]));
            if (j0 + 32 < NROW)
                __builtin_prefetch(arow[t] + j0 + 32 + pjj, 0, 0);
        }
        __syncthreads();

        // ---- consume: 2 M-tiles x 4 N-tiles of WMMA ----
        const int kb = j0 + half * 8;
        #pragma unroll
        for (int mt = 0; mt < 2; ++mt) {
            const unsigned short* ap = &sP[(mt * 16 + m) * 32 + half * 8];
            bf16x16 afrag = make_frag(*reinterpret_cast<const u32x4*>(ap),
                                      *reinterpret_cast<const u32x4*>(ap + 16));
            #pragma unroll
            for (int nt = 0; nt < 4; ++nt) {
                const unsigned short* bp =
                    hfcT + (size_t)(col0 + nt * 16 + m) * NROW + kb;
                bf16x16 bfrag = make_frag(*reinterpret_cast<const u32x4*>(bp),
                                          *reinterpret_cast<const u32x4*>(bp + 16));
                acc[mt][nt] = __builtin_amdgcn_wmma_f32_16x16x32_bf16(
                    false, afrag, false, bfrag, (short)0, acc[mt][nt], false, false);
            }
        }
        __syncthreads();
    }

    // ---- epilogue: out[row][col] = acc / rowsum[row] ----
    #pragma unroll
    for (int mt = 0; mt < 2; ++mt) {
        #pragma unroll
        for (int nt = 0; nt < 4; ++nt) {
            int col = col0 + nt * 16 + m;
            #pragma unroll
            for (int v = 0; v < 8; ++v) {
                int r = mt * 16 + half * 8 + v;
                out[(size_t)(row0 + r) * CDIM + col] = acc[mt][nt][v] * sInvL[r];
            }
        }
    }
}

// ---------------------------------------------------------------------------
// launcher
// ---------------------------------------------------------------------------
extern "C" void kernel_launch(void* const* d_in, const int* in_sizes, int n_in,
                              void* d_out, int out_size, void* d_ws, size_t ws_size,
                              hipStream_t stream) {
    const float* h    = (const float*)d_in[0];
    const float* adj  = (const float*)d_in[1];
    const float* W    = (const float*)d_in[2];
    const float* a_w  = (const float*)d_in[3];
    const float* a_b  = (const float*)d_in[4];
    const float* fc_w = (const float*)d_in[5];
    const float* fc_b = (const float*)d_in[6];
    float* out = (float*)d_out;

    char* ws = (char*)d_ws;
    const size_t OFF_S1   = 0;
    const size_t OFF_S2   = OFF_S1  + (size_t)NROW * 4;
    const size_t OFF_RM   = OFF_S2  + (size_t)NROW * 4;
    const size_t OFF_RS   = OFF_RM  + (size_t)NROW * 4;
    const size_t OFF_W1A  = OFF_RS  + (size_t)NROW * 4;
    const size_t OFF_W2A  = OFF_W1A + (size_t)CDIM * 4;
    const size_t OFF_HB   = OFF_W2A + (size_t)CDIM * 4;
    const size_t OFF_FCWT = OFF_HB   + (size_t)NROW * CDIM * 2;
    const size_t OFF_HFCT = OFF_FCWT + (size_t)CDIM * CDIM * 2;

    float* s1 = (float*)(ws + OFF_S1);
    float* s2 = (float*)(ws + OFF_S2);
    float* rm = (float*)(ws + OFF_RM);
    float* rs = (float*)(ws + OFF_RS);
    float* w1a = (float*)(ws + OFF_W1A);
    float* w2a = (float*)(ws + OFF_W2A);
    unsigned short* hb   = (unsigned short*)(ws + OFF_HB);
    unsigned short* fcwT = (unsigned short*)(ws + OFF_FCWT);
    unsigned short* hfcT = (unsigned short*)(ws + OFF_HFCT);

    k_prep_wa <<<128, 256, 0, stream>>>(W, a_w, w1a, w2a);                 // 1024 waves
    k_cvt_h   <<<(NROW * CDIM) / (256 * 4), 256, 0, stream>>>(h, (unsigned int*)hb);
    k_cvt_fcwT<<<(CDIM * CDIM) / 256, 256, 0, stream>>>(fc_w, fcwT);
    k_s12     <<<NROW / 8, 256, 0, stream>>>(h, w1a, w2a, s1, s2);         // wave/row
    k_hfc_gemm<<<dim3(NROW / 128, CDIM / 64), 256, 0, stream>>>(hb, fcwT, fc_b, hfcT);
    k_row_stats<<<NROW, 256, 0, stream>>>(adj, s1, s2, a_b, rm, rs);
    k_att_gemm<<<NROW / 32, 256, 0, stream>>>(adj, s1, s2, a_b, rm, rs, hfcT, out);
}
